// Pointnet_Backbone_19069654794727
// MI455X (gfx1250) — compile-verified, matrix operand order checked by reading, not run on hardware
//
#include <hip/hip_runtime.h>
#include <hip/hip_bf16.h>

// ---------------------------------------------------------------------------
// PointNet++ SSG backbone for MI455X (gfx1250, wave32, WMMA).
// GEMMs run on v_wmma_f32_16x16x32_f16 (f32 accumulate), 4 N-tiles register-
// blocked per wave (one A fragment feeds 4 WMMAs). All K-step loads issued
// into distinct fragments BEFORE the WMMA quartet so the scheduler can keep
// the next step's loads in flight behind the matrix ops.
// ---------------------------------------------------------------------------

typedef __attribute__((ext_vector_type(16))) _Float16 v16h;
typedef __attribute__((ext_vector_type(8)))  _Float16 h8;
typedef __attribute__((ext_vector_type(8)))  float    v8f;

#define BN_EPS 1e-5f
#define NSAMPLE 32

static __device__ __forceinline__ v16h cat16(h8 lo, h8 hi) {
  return __builtin_shufflevector(lo, hi, 0,1,2,3,4,5,6,7,8,9,10,11,12,13,14,15);
}

// ------------------------------ FPS ---------------------------------------
// One block (1024 threads) per batch. Coords cached in dynamic LDS (<=96KB of
// the 320KB WGP pool); per-thread min-dists live in registers; argmax via
// wave32 shuffle butterflies (2 barriers per iteration).
__global__ void fps_kernel(const float* __restrict__ xyz, int stride, int N,
                           int npoint, float* __restrict__ out_xyz) {
  extern __shared__ float sx[];          // N*3 floats
  __shared__ float wval[32];
  __shared__ int   widx[32];
  __shared__ int   sfar;
  const int b = blockIdx.x;
  const int t = threadIdx.x;
  const int nthr = blockDim.x;           // 1024
  const int lane = t & 31;
  const int wv = t >> 5;
  const int nw = nthr >> 5;
  const float* X = xyz + (size_t)b * N * stride;
  for (int n = t; n < N; n += nthr) {
    sx[n * 3 + 0] = X[(size_t)n * stride + 0];
    sx[n * 3 + 1] = X[(size_t)n * stride + 1];
    sx[n * 3 + 2] = X[(size_t)n * stride + 2];
  }
  float dist[8];
#pragma unroll
  for (int i = 0; i < 8; ++i) dist[i] = 1e10f;
  if (t == 0) sfar = 0;
  __syncthreads();
  for (int it = 0; it < npoint; ++it) {
    const int far = sfar;
    const float px = sx[far * 3 + 0];
    const float py = sx[far * 3 + 1];
    const float pz = sx[far * 3 + 2];
    if (t == 0) {
      float* o = out_xyz + ((size_t)b * npoint + it) * 3;
      o[0] = px; o[1] = py; o[2] = pz;
    }
    float best = -1.0f; int bi = 0;
    for (int i = 0, n = t; n < N; ++i, n += nthr) {
      float dx = sx[n * 3 + 0] - px;
      float dy = sx[n * 3 + 1] - py;
      float dz = sx[n * 3 + 2] - pz;
      float d = dx * dx + dy * dy + dz * dz;
      float m = fminf(dist[i], d);
      dist[i] = m;
      if (m > best) { best = m; bi = n; }   // ascending scan -> lowest idx on tie
    }
#pragma unroll
    for (int msk = 16; msk > 0; msk >>= 1) {
      float ov = __shfl_xor(best, msk, 32);
      int   oi = __shfl_xor(bi,   msk, 32);
      if (ov > best || (ov == best && oi < bi)) { best = ov; bi = oi; }
    }
    if (lane == 0) { wval[wv] = best; widx[wv] = bi; }
    __syncthreads();
    if (wv == 0) {
      float v = (lane < nw) ? wval[lane] : -1.0f;
      int  ii = (lane < nw) ? widx[lane] : 0x7fffffff;
#pragma unroll
      for (int msk = 16; msk > 0; msk >>= 1) {
        float ov = __shfl_xor(v,  msk, 32);
        int   oi = __shfl_xor(ii, msk, 32);
        if (ov > v || (ov == v && oi < ii)) { v = ov; ii = oi; }
      }
      if (lane == 0) sfar = ii;
    }
    __syncthreads();
  }
}

// --------------------------- Ball query -----------------------------------
// Reference top_k(-masked_idx) == first NSAMPLE indices within radius, padded
// with the first hit. Thread per center, early exit (dense clouds hit fast).
__global__ void ball_query_kernel(const float* __restrict__ centers,
                                  const float* __restrict__ xyz, int stride,
                                  int N, int S, int total, float r2,
                                  int* __restrict__ idx) {
  int g = blockIdx.x * blockDim.x + threadIdx.x;
  if (g >= total) return;
  int b = g / S;
  const float* c = centers + (size_t)g * 3;
  const float cx = c[0], cy = c[1], cz = c[2];
  const float* X = xyz + (size_t)b * N * stride;
  int* out = idx + (size_t)g * NSAMPLE;
  int cnt = 0, first = 0;
  for (int n = 0; n < N && cnt < NSAMPLE; ++n) {
    float dx = X[(size_t)n * stride + 0] - cx;
    float dy = X[(size_t)n * stride + 1] - cy;
    float dz = X[(size_t)n * stride + 2] - cz;
    if (dx * dx + dy * dy + dz * dz <= r2) {
      if (cnt == 0) first = n;
      out[cnt++] = n;
    }
  }
  for (int k = cnt; k < NSAMPLE; ++k) out[k] = first;
}

// ------------------------- Group / gather to f16 A -------------------------
__global__ void gather_kernel(const float* __restrict__ xyz, int xstride,
                              const float* __restrict__ feats, int fstride, int Cfeat,
                              const float* __restrict__ centers,
                              const int* __restrict__ idx,
                              int S, int N, int KP, int M,
                              _Float16* __restrict__ A) {
  int m = blockIdx.x * blockDim.x + threadIdx.x;
  if (m >= M) return;
  int bs = m >> 5;            // NSAMPLE == 32
  int b = bs / S;
  int n = idx[m];
  const float* c = centers + (size_t)bs * 3;
  const float* X = xyz + ((size_t)b * N + n) * xstride;
  _Float16* a = A + (size_t)m * KP;
  a[0] = (_Float16)(X[0] - c[0]);
  a[1] = (_Float16)(X[1] - c[1]);
  a[2] = (_Float16)(X[2] - c[2]);
  const float* F = feats + ((size_t)b * N + n) * fstride;
  for (int ci = 0; ci < Cfeat; ++ci) a[3 + ci] = (_Float16)F[ci];
  for (int k = 3 + Cfeat; k < KP; ++k) a[k] = (_Float16)0.0f;
}

// -------------------- Weight f32 (N x K) -> f16 (N x KP) -------------------
__global__ void wprep_kernel(const float* __restrict__ W, int N, int K, int KP,
                             _Float16* __restrict__ Wh) {
  int i = blockIdx.x * blockDim.x + threadIdx.x;
  if (i >= N * KP) return;
  int n = i / KP, k = i % KP;
  Wh[i] = (k < K) ? (_Float16)W[n * K + k] : (_Float16)0.0f;
}

__global__ void f32_to_f16_kernel(const float* __restrict__ in,
                                  _Float16* __restrict__ out, int n) {
  int i = blockIdx.x * blockDim.x + threadIdx.x;
  if (i < n) out[i] = (_Float16)in[i];
}

// ----------------------------- WMMA GEMM ----------------------------------
// H[M x Nc] (f16) = A[M x KA] (f16) * Wh[Nc x KA]^T, f32 accumulate.
// Block = 128 threads = 4 waves. Wave w owns M-tile (blockIdx.x*4+w)*16 and
// FOUR 16-wide N tiles (blockIdx.y*64 .. +63). All 10 loads of a K-step are
// issued into distinct fragments before the 4 WMMAs so subsequent loads can
// stay in flight behind the matrix pipe. BN per-channel sum/sumsq are
// block-reduced in LDS, then 2 atomics per channel per block.
__global__ void wmma_gemm_stats_kernel(const _Float16* __restrict__ A, int KA,
                                       const _Float16* __restrict__ Wh,
                                       _Float16* __restrict__ H, int Nc,
                                       float* __restrict__ gsum,
                                       float* __restrict__ gsumsq) {
  const int lane = threadIdx.x & 31;
  const int wave = threadIdx.x >> 5;
  const int m0 = (blockIdx.x * 4 + wave) * 16;
  const int n0 = blockIdx.y * 64;
  const int r = lane & 15;
  const int koff = (lane >> 4) * 8;   // f16 A frag: lane<16 K0-7/K16-23, lane>=16 K8-15/K24-31
  const _Float16* arow = A  + (size_t)(m0 + r) * KA + koff;
  const _Float16* brow = Wh + (size_t)(n0 + r) * KA + koff;
  const size_t bstr = (size_t)16 * KA;       // next 16-channel tile of W
  v8f acc0 = {}, acc1 = {}, acc2 = {}, acc3 = {};
  for (int k0 = 0; k0 < KA; k0 += 32) {
    __builtin_prefetch(arow + k0 + 64, 0, 1);   // global_prefetch_b8
    const _Float16* ap = arow + k0;
    const _Float16* bp = brow + k0;
    // Load phase: 10 independent b128 loads.
    h8 alo = *(const h8*)(ap);
    h8 ahi = *(const h8*)(ap + 16);
    h8 b0l = *(const h8*)(bp);
    h8 b0h = *(const h8*)(bp + 16);
    h8 b1l = *(const h8*)(bp + bstr);
    h8 b1h = *(const h8*)(bp + bstr + 16);
    h8 b2l = *(const h8*)(bp + 2 * bstr);
    h8 b2h = *(const h8*)(bp + 2 * bstr + 16);
    h8 b3l = *(const h8*)(bp + 3 * bstr);
    h8 b3h = *(const h8*)(bp + 3 * bstr + 16);
    // Math phase: one A fragment feeds four WMMAs.
    v16h a = cat16(alo, ahi);
    acc0 = __builtin_amdgcn_wmma_f32_16x16x32_f16(false, a, false, cat16(b0l, b0h),
                                                  (short)0, acc0, false, false);
    acc1 = __builtin_amdgcn_wmma_f32_16x16x32_f16(false, a, false, cat16(b1l, b1h),
                                                  (short)0, acc1, false, false);
    acc2 = __builtin_amdgcn_wmma_f32_16x16x32_f16(false, a, false, cat16(b2l, b2h),
                                                  (short)0, acc2, false, false);
    acc3 = __builtin_amdgcn_wmma_f32_16x16x32_f16(false, a, false, cat16(b3l, b3h),
                                                  (short)0, acc3, false, false);
  }
  v8f accv[4] = {acc0, acc1, acc2, acc3};
  // Epilogue: D layout lane<16 -> M=r(0..7), lane>=16 -> M=8+r; N = lane&15.
  __shared__ float ps1[64][8];
  __shared__ float ps2[64][8];
  const int slot = wave * 2 + (lane >> 4);
  const int mbase = m0 + ((lane >> 4) << 3);
#pragma unroll
  for (int tt = 0; tt < 4; ++tt) {
    const int cl = tt * 16 + (lane & 15);   // local channel 0..63
    const int n = n0 + cl;
    float s1 = 0.0f, s2 = 0.0f;
#pragma unroll
    for (int rr = 0; rr < 8; ++rr) {
      float v = accv[tt][rr];
      H[(size_t)(mbase + rr) * Nc + n] = (_Float16)v;
      s1 += v; s2 += v * v;
    }
    ps1[cl][slot] = s1;
    ps2[cl][slot] = s2;
  }
  __syncthreads();
  if (threadIdx.x < 64) {
    float t1 = 0.0f, t2 = 0.0f;
    for (int q = 0; q < 8; ++q) { t1 += ps1[threadIdx.x][q]; t2 += ps2[threadIdx.x][q]; }
    atomicAdd(&gsum[n0 + threadIdx.x], t1);
    atomicAdd(&gsumsq[n0 + threadIdx.x], t2);
  }
}

// --------------------------- BN + ReLU (in place) ---------------------------
__global__ void bn_relu_kernel(_Float16* __restrict__ H,
                               const float* __restrict__ gsum,
                               const float* __restrict__ gsumsq,
                               const float* __restrict__ gamma,
                               const float* __restrict__ beta,
                               int Nc, long long M) {
  long long i = (long long)blockIdx.x * blockDim.x + threadIdx.x;
  if (i >= M * Nc) return;
  int c = (int)(i % Nc);
  float inv_cnt = 1.0f / (float)M;
  float mean = gsum[c] * inv_cnt;
  float var  = gsumsq[c] * inv_cnt - mean * mean;
  float v = ((float)H[i] - mean) * rsqrtf(var + BN_EPS) * gamma[c] + beta[c];
  H[i] = (_Float16)fmaxf(v, 0.0f);
}

// ----------------- BN + ReLU + maxpool over ns (last layer) ----------------
__global__ void bn_maxpool_kernel(const _Float16* __restrict__ H,
                                  const float* __restrict__ gsum,
                                  const float* __restrict__ gsumsq,
                                  const float* __restrict__ gamma,
                                  const float* __restrict__ beta,
                                  int Nc, int S, long long M,
                                  float* __restrict__ pooled,     // (B,S,C)
                                  float* __restrict__ feat_out) { // (B,C,S)
  int total = (int)(M >> 5) * Nc;   // B*S*Nc
  int i = blockIdx.x * blockDim.x + threadIdx.x;
  if (i >= total) return;
  int c = i % Nc;
  int bs = i / Nc;
  int b = bs / S, s = bs % S;
  float inv_cnt = 1.0f / (float)M;
  float mean = gsum[c] * inv_cnt;
  float var  = gsumsq[c] * inv_cnt - mean * mean;
  float scale = rsqrtf(var + BN_EPS) * gamma[c];
  float shift = beta[c];
  const _Float16* hp = H + ((size_t)bs * NSAMPLE) * Nc + c;
  float mx = -1e30f;
  for (int j = 0; j < NSAMPLE; ++j) {
    float v = fmaxf(((float)hp[(size_t)j * Nc] - mean) * scale + shift, 0.0f);
    mx = fmaxf(mx, v);
  }
  pooled[(size_t)bs * Nc + c] = mx;
  feat_out[((size_t)b * Nc + c) * S + s] = mx;
}

// -------------------------- Final conv (WMMA) ------------------------------
// out[b, n, s] = sum_c feat3[b,s,c] * wf[n,c] + bf[n];  M=1024, K=N=256.
__global__ void wmma_gemm_final_kernel(const _Float16* __restrict__ A,
                                       const _Float16* __restrict__ Wh,
                                       const float* __restrict__ bias,
                                       float* __restrict__ out) {
  const int KA = 256;
  const int lane = threadIdx.x & 31;
  const int wave = threadIdx.x >> 5;
  const int m0 = (blockIdx.x * 4 + wave) * 16;
  const int n0 = blockIdx.y * 64;
  const int r = lane & 15;
  const int koff = (lane >> 4) * 8;
  const _Float16* arow = A  + (size_t)(m0 + r) * KA + koff;
  const _Float16* brow = Wh + (size_t)(n0 + r) * KA + koff;
  const size_t bstr = (size_t)16 * KA;
  v8f acc0 = {}, acc1 = {}, acc2 = {}, acc3 = {};
  for (int k0 = 0; k0 < KA; k0 += 32) {
    const _Float16* ap = arow + k0;
    const _Float16* bp = brow + k0;
    h8 alo = *(const h8*)(ap);
    h8 ahi = *(const h8*)(ap + 16);
    h8 b0l = *(const h8*)(bp);
    h8 b0h = *(const h8*)(bp + 16);
    h8 b1l = *(const h8*)(bp + bstr);
    h8 b1h = *(const h8*)(bp + bstr + 16);
    h8 b2l = *(const h8*)(bp + 2 * bstr);
    h8 b2h = *(const h8*)(bp + 2 * bstr + 16);
    h8 b3l = *(const h8*)(bp + 3 * bstr);
    h8 b3h = *(const h8*)(bp + 3 * bstr + 16);
    v16h a = cat16(alo, ahi);
    acc0 = __builtin_amdgcn_wmma_f32_16x16x32_f16(false, a, false, cat16(b0l, b0h),
                                                  (short)0, acc0, false, false);
    acc1 = __builtin_amdgcn_wmma_f32_16x16x32_f16(false, a, false, cat16(b1l, b1h),
                                                  (short)0, acc1, false, false);
    acc2 = __builtin_amdgcn_wmma_f32_16x16x32_f16(false, a, false, cat16(b2l, b2h),
                                                  (short)0, acc2, false, false);
    acc3 = __builtin_amdgcn_wmma_f32_16x16x32_f16(false, a, false, cat16(b3l, b3h),
                                                  (short)0, acc3, false, false);
  }
  v8f accv[4] = {acc0, acc1, acc2, acc3};
  const int mbase = m0 + ((lane >> 4) << 3);
#pragma unroll
  for (int tt = 0; tt < 4; ++tt) {
    const int n = n0 + tt * 16 + (lane & 15);
    const float bn = bias[n];
#pragma unroll
    for (int rr = 0; rr < 8; ++rr) {
      int m = mbase + rr;
      int b = m >> 7, s = m & 127;
      out[((size_t)b * 256 + n) * 128 + s] = accv[tt][rr] + bn;
    }
  }
}

// ---------------------------------------------------------------------------
// Host-side orchestration
// ---------------------------------------------------------------------------
static void run_layer(const float* W, const float* g, const float* be,
                      int K, int KA, int N, long long M,
                      const _Float16* Ain, _Float16* Hout,
                      _Float16* Wh, float* gsum,
                      bool last, int S, float* pooled, float* feat_out,
                      hipStream_t stream) {
  wprep_kernel<<<(N * KA + 255) / 256, 256, 0, stream>>>(W, N, K, KA, Wh);
  hipMemsetAsync(gsum, 0, 512 * sizeof(float), stream);
  dim3 grid((unsigned)(M / 64), (unsigned)(N / 64));
  wmma_gemm_stats_kernel<<<grid, 128, 0, stream>>>(Ain, KA, Wh, Hout, N,
                                                   gsum, gsum + 256);
  if (last) {
    int total = (int)(M >> 5) * N;
    bn_maxpool_kernel<<<(total + 255) / 256, 256, 0, stream>>>(
        Hout, gsum, gsum + 256, g, be, N, S, M, pooled, feat_out);
  } else {
    long long total = M * N;
    bn_relu_kernel<<<(unsigned)((total + 255) / 256), 256, 0, stream>>>(
        Hout, gsum, gsum + 256, g, be, N, M);
  }
}

extern "C" void kernel_launch(void* const* d_in, const int* in_sizes, int n_in,
                              void* d_out, int out_size, void* d_ws, size_t ws_size,
                              hipStream_t stream) {
  (void)in_sizes; (void)n_in; (void)out_size; (void)ws_size;
  const int B = 8, N0 = 8192;
  const int S1 = 2048, S2 = 512, S3 = 128;
  const long long M1 = (long long)B * S1 * NSAMPLE;   // 524288
  const long long M2 = (long long)B * S2 * NSAMPLE;   // 131072
  const long long M3 = (long long)B * S3 * NSAMPLE;   // 32768

  const float* pc = (const float*)d_in[0];
  const float* wf = (const float*)d_in[28];
  const float* bf = (const float*)d_in[29];

  // d_out layout (flat f32, tuple order)
  float* out = (float*)d_out;
  float* xyz1 = out;                       // (8,2048,3)
  float* xyz2 = out + 49152;               // (8,512,3)
  float* xyz3 = out + 61440;               // (8,128,3)
  float* f1   = out + 64512;               // (8,128,2048)
  float* f2   = out + 2161664;             // (8,256,512)
  float* f3   = out + 3210240;             // (8,256,128)
  float* fout = out + 3472384;             // (8,256,128)

  // Workspace carve-up (256B aligned)
  size_t off = 0;
  auto take = [&](size_t bytes) {
    size_t o = off; off = (off + bytes + 255) & ~(size_t)255; return o;
  };
  char* ws = (char*)d_ws;
  _Float16* bufA   = (_Float16*)(ws + take((size_t)M1 * 64  * 2));  // 64 MB
  _Float16* bufB   = (_Float16*)(ws + take((size_t)M1 * 128 * 2));  // 128 MB
  int*      idxbuf = (int*)     (ws + take((size_t)M1 * 4));
  float* pooled1   = (float*)   (ws + take((size_t)B * S1 * 128 * 4));
  float* pooled2   = (float*)   (ws + take((size_t)B * S2 * 256 * 4));
  float* pooled3   = (float*)   (ws + take((size_t)B * S3 * 256 * 4));
  _Float16* Wh     = (_Float16*)(ws + take((size_t)256 * 288 * 2));
  float* gsum      = (float*)   (ws + take(512 * 4));

  // ---------------- SA module 1: N=8192 -> S=2048, r=0.3, [6,64,64,128] ----
  fps_kernel<<<B, 1024, (size_t)N0 * 3 * sizeof(float), stream>>>(pc, 6, N0, S1, xyz1);
  ball_query_kernel<<<(B * S1 + 255) / 256, 256, 0, stream>>>(
      xyz1, pc, 6, N0, S1, B * S1, 0.09f, idxbuf);
  gather_kernel<<<(unsigned)((M1 + 255) / 256), 256, 0, stream>>>(
      pc, 6, pc + 3, 6, 3, xyz1, idxbuf, S1, N0, 32, (int)M1, bufA);
  run_layer((const float*)d_in[1], (const float*)d_in[2], (const float*)d_in[3],
            6, 32, 64, M1, bufA, bufB, Wh, gsum, false, 0, nullptr, nullptr, stream);
  run_layer((const float*)d_in[4], (const float*)d_in[5], (const float*)d_in[6],
            64, 64, 64, M1, bufB, bufA, Wh, gsum, false, 0, nullptr, nullptr, stream);
  run_layer((const float*)d_in[7], (const float*)d_in[8], (const float*)d_in[9],
            64, 64, 128, M1, bufA, bufB, Wh, gsum, true, S1, pooled1, f1, stream);

  // ---------------- SA module 2: 2048 -> 512, r=0.5, [131,128,128,256] -----
  fps_kernel<<<B, 1024, (size_t)S1 * 3 * sizeof(float), stream>>>(xyz1, 3, S1, S2, xyz2);
  ball_query_kernel<<<(B * S2 + 255) / 256, 256, 0, stream>>>(
      xyz2, xyz1, 3, S1, S2, B * S2, 0.25f, idxbuf);
  gather_kernel<<<(unsigned)((M2 + 255) / 256), 256, 0, stream>>>(
      xyz1, 3, pooled1, 128, 128, xyz2, idxbuf, S2, S1, 160, (int)M2, bufA);
  run_layer((const float*)d_in[10], (const float*)d_in[11], (const float*)d_in[12],
            131, 160, 128, M2, bufA, bufB, Wh, gsum, false, 0, nullptr, nullptr, stream);
  run_layer((const float*)d_in[13], (const float*)d_in[14], (const float*)d_in[15],
            128, 128, 128, M2, bufB, bufA, Wh, gsum, false, 0, nullptr, nullptr, stream);
  run_layer((const float*)d_in[16], (const float*)d_in[17], (const float*)d_in[18],
            128, 128, 256, M2, bufA, bufB, Wh, gsum, true, S2, pooled2, f2, stream);

  // ---------------- SA module 3: 512 -> 128, r=0.7, [259,256,256,256] ------
  fps_kernel<<<B, 1024, (size_t)S2 * 3 * sizeof(float), stream>>>(xyz2, 3, S2, S3, xyz3);
  ball_query_kernel<<<(B * S3 + 255) / 256, 256, 0, stream>>>(
      xyz3, xyz2, 3, S2, S3, B * S3, 0.49f, idxbuf);
  gather_kernel<<<(unsigned)((M3 + 255) / 256), 256, 0, stream>>>(
      xyz2, 3, pooled2, 256, 256, xyz3, idxbuf, S3, S2, 288, (int)M3, bufA);
  run_layer((const float*)d_in[19], (const float*)d_in[20], (const float*)d_in[21],
            259, 288, 256, M3, bufA, bufB, Wh, gsum, false, 0, nullptr, nullptr, stream);
  run_layer((const float*)d_in[22], (const float*)d_in[23], (const float*)d_in[24],
            256, 256, 256, M3, bufB, bufA, Wh, gsum, false, 0, nullptr, nullptr, stream);
  run_layer((const float*)d_in[25], (const float*)d_in[26], (const float*)d_in[27],
            256, 256, 256, M3, bufA, bufB, Wh, gsum, true, S3, pooled3, f3, stream);

  // ---------------- Final 1x1 conv: (8,128,256) x (256,256) + bias --------
  f32_to_f16_kernel<<<(B * S3 * 256 + 255) / 256, 256, 0, stream>>>(
      pooled3, bufA, B * S3 * 256);
  wprep_kernel<<<(256 * 256 + 255) / 256, 256, 0, stream>>>(wf, 256, 256, 256, Wh);
  wmma_gemm_final_kernel<<<dim3(16, 4), 128, 0, stream>>>(bufA, Wh, bf, fout);
}
